// TriTotnet_52149492908333
// MI455X (gfx1250) — matrix operand order repaired and checked by reading, not compiled, not run on hardware
//
#include <hip/hip_runtime.h>
#include <hip/hip_bf16.h>
#include <math.h>

// ---------------- problem constants ----------------
#define B_    4096
#define V_    3
#define P_    40
#define DIN_  128
#define SH_   128
#define SO_   64
#define H1_   2048
#define H2_   512
#define OUT_  256
#define BV_   (B_ * V_)          // 12288 rows
#define PSO_  (P_ * SO_)         // 2560

typedef __attribute__((ext_vector_type(16))) __bf16 v16bf;
typedef __attribute__((ext_vector_type(8)))  __bf16 v8bf;   // 16-byte chunk
typedef __attribute__((ext_vector_type(8)))  float  v8f;

// ---------------- helpers ----------------
__device__ __forceinline__ unsigned pack_bf2(float a, float b) {
  union { __bf16 h[2]; unsigned u; } un;
  un.h[0] = (__bf16)a;
  un.h[1] = (__bf16)b;
  return un.u;
}

// A-matrix 16x32 bf16 fragment from row-major LDS (leading dim ld):
// lanes 0-15 -> M=lane, K 0..15 ; lanes 16-31 -> M=lane-16, K 16..31
__device__ __forceinline__ v16bf frag_a(const __bf16* base, int ld) {
  const int lane = threadIdx.x & 31;
  const int row  = lane & 15;
  const int kb   = (lane >> 4) << 4;     // 0 or 16
  const __bf16* p = base + row * ld;
  v16bf f;
#pragma unroll
  for (int e = 0; e < 8; ++e) {
    f[e]     = p[kb + e];
    f[e + 8] = p[kb + 8 + e];
  }
  return f;
}

// B-matrix fragment from LDS that was staged in fragment order:
// tile block of 512 bf16, lane-contiguous: [lane][e] with
//   lane = N + 16*(K>=16), e = K & 15   (32x16 tile)
__device__ __forceinline__ v16bf frag_b_sw(const __bf16* tilebase) {
  const int lane = threadIdx.x & 31;
  return *(const v16bf*)(tilebase + lane * 16);
}

__device__ __forceinline__ v8f wmma_bf16(v16bf a, v16bf b, v8f c) {
  return __builtin_amdgcn_wmma_f32_16x16x32_bf16(false, a, false, b, (short)0, c,
                                                 false, false);
}

// ---------------- Kernel 1: per-phone subnet (grouped 2-layer MLP) ----------------
// grid: (BV/16, P), block: 256
__global__ __launch_bounds__(256) void subnet_kernel(
    const float* __restrict__ x,  const float* __restrict__ W1,
    const float* __restrict__ b1, const float* __restrict__ W2,
    const float* __restrict__ b2, __bf16* __restrict__ act0) {
  __shared__ alignas(32) __bf16 xA[16 * DIN_];   //  4 KB, row-major
  __shared__ alignas(32) __bf16 w1[DIN_ * SH_];  // 32 KB, swizzled tiles (kt*8+ct)
  __shared__ alignas(32) __bf16 hs[16 * SH_];    //  4 KB, row-major
  __shared__ alignas(32) __bf16 w2[SH_ * SO_];   // 16 KB, swizzled tiles (kt*4+ct)

  const int r0 = blockIdx.x * 16;
  const int p  = blockIdx.y;
  const int t  = threadIdx.x;

  // stage x tile (f32 -> bf16 pairs), row-major
  for (int i = t; i < 512; i += 256) {           // 16 rows * 32 groups of 4
    const int r = i >> 5, c4 = (i & 31) * 4;
    const float4 v = *(const float4*)(x + (size_t)(r0 + r) * (P_ * DIN_) + p * DIN_ + c4);
    uint2 u;
    u.x = pack_bf2(v.x, v.y);
    u.y = pack_bf2(v.z, v.w);
    ((uint2*)xA)[(r * DIN_ + c4) >> 2] = u;
  }
  // stage W1 (128x128) swizzled: tile = (k>>5)*8 + (c>>4)
  {
    const float* W1p = W1 + (size_t)p * DIN_ * SH_;
    for (int i = t; i < 8192; i += 256) {        // 64 k-pairs * 128 cols
      const int kp = i >> 7, c = i & 127;
      const int k = kp * 2;
      const float f0 = W1p[(size_t)k * SH_ + c];
      const float f1 = W1p[(size_t)(k + 1) * SH_ + c];
      const int kin = k & 31;
      const int lane = (c & 15) + 16 * (kin >> 4);
      const int e = kin & 15;
      const int tile = (k >> 5) * 8 + (c >> 4);
      ((unsigned*)w1)[(tile * 512 + lane * 16 + e) >> 1] = pack_bf2(f0, f1);
    }
  }
  // stage W2 (128x64) swizzled: tile = (k>>5)*4 + (c>>4)
  {
    const float* W2p = W2 + (size_t)p * SH_ * SO_;
    for (int i = t; i < 4096; i += 256) {        // 64 k-pairs * 64 cols
      const int kp = i >> 6, c = i & 63;
      const int k = kp * 2;
      const float f0 = W2p[(size_t)k * SO_ + c];
      const float f1 = W2p[(size_t)(k + 1) * SO_ + c];
      const int kin = k & 31;
      const int lane = (c & 15) + 16 * (kin >> 4);
      const int e = kin & 15;
      const int tile = (k >> 5) * 4 + (c >> 4);
      ((unsigned*)w2)[(tile * 512 + lane * 16 + e) >> 1] = pack_bf2(f0, f1);
    }
  }
  __syncthreads();

  const int wave = t >> 5, lane = t & 31;

  // GEMM1: h = relu(x @ W1 + b1); wave w owns SH cols [16w, 16w+16)
  {
    v8f acc = {};
#pragma unroll
    for (int kt = 0; kt < DIN_ / 32; ++kt) {
      v16bf a  = frag_a(xA + kt * 32, DIN_);
      v16bf bm = frag_b_sw(w1 + (kt * 8 + wave) * 512);
      acc = wmma_bf16(a, bm, acc);
    }
    const int col  = wave * 16 + (lane & 15);
    const float bb = b1[p * SH_ + col];
    const int mofs = (lane >> 4) * 8;
#pragma unroll
    for (int j = 0; j < 8; ++j) {
      float v = acc[j] + bb;
      v = v > 0.0f ? v : 0.0f;
      hs[(mofs + j) * SH_ + col] = (__bf16)v;
    }
  }
  __syncthreads();

  // GEMM2: s = h @ W2 + b2; waves 0..3 own SO cols (wave-uniform branch)
  if (wave < 4) {
    v8f acc = {};
#pragma unroll
    for (int kt = 0; kt < SH_ / 32; ++kt) {
      v16bf a  = frag_a(hs + kt * 32, SH_);
      v16bf bm = frag_b_sw(w2 + (kt * 4 + wave) * 512);
      acc = wmma_bf16(a, bm, acc);
    }
    const int col  = wave * 16 + (lane & 15);
    const float bb = b2[p * SO_ + col];
    const int mofs = (lane >> 4) * 8;
#pragma unroll
    for (int j = 0; j < 8; ++j) {
      const float v = acc[j] + bb;
      act0[(size_t)(r0 + mofs + j) * PSO_ + p * SO_ + col] = (__bf16)v;
    }
  }
}

// ---------------- Kernel 2: generic GEMM + bias (+ReLU), bf16 act, f32 weights ----
// grid: (M/64, N/64), block: 256. Each wave: 16x32 (2 accumulators).
template <int RELU>
__global__ __launch_bounds__(256) void gemm_bias_kernel(
    const __bf16* __restrict__ A, const float* __restrict__ W,
    const float* __restrict__ bias, __bf16* __restrict__ C,
    int M, int K, int N) {
  __shared__ alignas(32) __bf16 As[64 * 32];   // 4 KB, row-major
  __shared__ alignas(32) __bf16 Ws[32 * 64];   // 4 KB, swizzled tiles 0..3

  const int r0 = blockIdx.x * 64;
  const int c0 = blockIdx.y * 64;
  const int t = threadIdx.x, wave = t >> 5, lane = t & 31;
  const int wr = (wave >> 1) * 16;   // 0,16,32,48
  const int wc = wave & 1;           // tile pair: 2*wc, 2*wc+1

  v8f acc0 = {}, acc1 = {};
  for (int k0 = 0; k0 < K; k0 += 32) {
    // A panel 64x32: one 16-byte chunk per thread
    {
      const int row = t >> 2, ch = (t & 3) * 8;
      ((v8bf*)As)[t] = *(const v8bf*)(A + (size_t)(r0 + row) * K + k0 + ch);
    }
    // W panel 32x64 -> swizzled fragment order
    for (int i = t; i < 1024; i += 256) {        // 16 k-pairs * 64 cols
      const int kp = i >> 6, c = i & 63;
      const int k = kp * 2;
      const float f0 = W[(size_t)(k0 + k) * N + c0 + c];
      const float f1 = W[(size_t)(k0 + k + 1) * N + c0 + c];
      const int lane2 = (c & 15) + 16 * (k >> 4);
      const int e = k & 15;
      const int tile = c >> 4;
      ((unsigned*)Ws)[(tile * 512 + lane2 * 16 + e) >> 1] = pack_bf2(f0, f1);
    }
    if (k0 + 32 < K)
      __builtin_prefetch(W + (size_t)(k0 + 32) * N + c0 + (t & 63), 0, 1);
    __syncthreads();
    v16bf a   = frag_a(As + wr * 32, 32);
    v16bf bm0 = frag_b_sw(Ws + (2 * wc) * 512);
    v16bf bm1 = frag_b_sw(Ws + (2 * wc + 1) * 512);
    acc0 = wmma_bf16(a, bm0, acc0);
    acc1 = wmma_bf16(a, bm1, acc1);
    __syncthreads();
  }

  const int colA = c0 + wc * 32 + (lane & 15);
  const int colB = colA + 16;
  const float bA = bias[colA], bB = bias[colB];
  const int mofs = (lane >> 4) * 8;
#pragma unroll
  for (int j = 0; j < 8; ++j) {
    const int row = r0 + wr + mofs + j;
    float v0 = acc0[j] + bA;
    float v1 = acc1[j] + bB;
    if (RELU) { v0 = v0 > 0.0f ? v0 : 0.0f; v1 = v1 > 0.0f ? v1 : 0.0f; }
    C[(size_t)row * N + colA] = (__bf16)v0;
    C[(size_t)row * N + colB] = (__bf16)v1;
  }
}

// ---------------- Kernel 3: head layer 3 + L2 normalize + scatter ----------------
// grid: BV/16, block: 256. Full 16x256 tile per block, K = 512.
__global__ __launch_bounds__(256) void head3_norm_kernel(
    const __bf16* __restrict__ A, const float* __restrict__ W,
    const float* __restrict__ bias, float* __restrict__ out) {
  __shared__ alignas(32) __bf16 As[16 * 32];     //  1 KB, row-major
  __shared__ alignas(32) __bf16 Ws[32 * OUT_];   // 16 KB, swizzled tiles 0..15
  __shared__ float  zb[16 * OUT_];               // 16 KB
  __shared__ float  psum[16 * 16];
  __shared__ float  scale[16];

  const int r0 = blockIdx.x * 16;
  const int t = threadIdx.x, wave = t >> 5, lane = t & 31;

  v8f acc0 = {}, acc1 = {};
  for (int k0 = 0; k0 < H2_; k0 += 32) {
    // A panel 16x32: 64 16-byte chunks
    for (int i = t; i < 64; i += 256) {
      const int row = i >> 2, ch = (i & 3) * 8;
      ((v8bf*)As)[i] = *(const v8bf*)(A + (size_t)(r0 + row) * H2_ + k0 + ch);
    }
    // W panel 32x256 -> swizzled fragment order (16 tiles)
    for (int i = t; i < 4096; i += 256) {        // 16 k-pairs * 256 cols
      const int kp = i >> 8, c = i & 255;
      const int k = kp * 2;
      const float f0 = W[(size_t)(k0 + k) * OUT_ + c];
      const float f1 = W[(size_t)(k0 + k + 1) * OUT_ + c];
      const int lane2 = (c & 15) + 16 * (k >> 4);
      const int e = k & 15;
      const int tile = c >> 4;
      ((unsigned*)Ws)[(tile * 512 + lane2 * 16 + e) >> 1] = pack_bf2(f0, f1);
    }
    if (k0 + 32 < H2_)
      __builtin_prefetch(W + (size_t)(k0 + 32) * OUT_ + t, 0, 1);
    __syncthreads();
    v16bf a   = frag_a(As, 32);
    v16bf bm0 = frag_b_sw(Ws + (2 * wave) * 512);
    v16bf bm1 = frag_b_sw(Ws + (2 * wave + 1) * 512);
    acc0 = wmma_bf16(a, bm0, acc0);
    acc1 = wmma_bf16(a, bm1, acc1);
    __syncthreads();
  }

  // bias, park f32 results in LDS
  {
    const int col  = wave * 32 + (lane & 15);
    const int mofs = (lane >> 4) * 8;
    const float bA = bias[col], bB = bias[col + 16];
#pragma unroll
    for (int j = 0; j < 8; ++j) {
      zb[(mofs + j) * OUT_ + col]      = acc0[j] + bA;
      zb[(mofs + j) * OUT_ + col + 16] = acc1[j] + bB;
    }
  }
  __syncthreads();

  // per-row sum of squares: 16 threads per row
  {
    const int row = t >> 4, seg = t & 15;
    float s = 0.0f;
#pragma unroll
    for (int c = 0; c < 16; ++c) {
      const float v = zb[row * OUT_ + seg * 16 + c];
      s += v * v;
    }
    psum[row * 16 + seg] = s;
  }
  __syncthreads();
  if (t < 16) {
    float s = 0.0f;
#pragma unroll
    for (int i = 0; i < 16; ++i) s += psum[t * 16 + i];
    const float n = sqrtf(s);
    scale[t] = 16.0f / fmaxf(n, 1e-12f);   // sqrt(OUT) = 16
  }
  __syncthreads();

  // scatter: row rr = b*V + v  ->  out[v][b][col]
  for (int i = t; i < 16 * OUT_; i += 256) {
    const int r = i >> 8, c = i & 255;
    const int rr = r0 + r;
    const int b = rr / V_, v = rr % V_;
    out[(size_t)v * B_ * OUT_ + (size_t)b * OUT_ + c] = zb[i] * scale[r];
  }
}

// ---------------- host: launch pipeline ----------------
extern "C" void kernel_launch(void* const* d_in, const int* in_sizes, int n_in,
                              void* d_out, int out_size, void* d_ws, size_t ws_size,
                              hipStream_t stream) {
  const float* x   = (const float*)d_in[0];
  const float* W1  = (const float*)d_in[1];
  const float* b1  = (const float*)d_in[2];
  const float* W2  = (const float*)d_in[3];
  const float* b2  = (const float*)d_in[4];
  const float* Wf1 = (const float*)d_in[5];
  const float* bf1 = (const float*)d_in[6];
  const float* Wf2 = (const float*)d_in[7];
  const float* bf2 = (const float*)d_in[8];
  const float* Wf3 = (const float*)d_in[9];
  const float* bf3 = (const float*)d_in[10];
  float* out = (float*)d_out;

  // bf16 activation scratch
  char* ws = (char*)d_ws;
  __bf16* act0 = (__bf16*)ws;                                    // [BV, 2560]
  __bf16* act1 = (__bf16*)(ws + (size_t)BV_ * PSO_ * 2);         // [BV, 2048]
  __bf16* act2 = (__bf16*)(ws + (size_t)BV_ * (PSO_ + H1_) * 2); // [BV, 512]
  (void)in_sizes; (void)n_in; (void)out_size; (void)ws_size;

  // 1) per-phone subnets -> flat [BV, P*SO]
  subnet_kernel<<<dim3(BV_ / 16, P_), 256, 0, stream>>>(x, W1, b1, W2, b2, act0);
  // 2) z1 = relu(flat @ Wf1 + bf1)   [BV, 2048]
  gemm_bias_kernel<1><<<dim3(BV_ / 64, H1_ / 64), 256, 0, stream>>>(
      act0, Wf1, bf1, act1, BV_, PSO_, H1_);
  // 3) z2 = relu(z1 @ Wf2 + bf2)     [BV, 512]
  gemm_bias_kernel<1><<<dim3(BV_ / 64, H2_ / 64), 256, 0, stream>>>(
      act1, Wf2, bf2, act2, BV_, H1_, H2_);
  // 4) z3 = z2 @ Wf3 + bf3, L2-normalize * 16, scatter per-view
  head3_norm_kernel<<<dim3(BV_ / 16), 256, 0, stream>>>(act2, Wf3, bf3, out);
}
